// SourceAttention_12292196401328
// MI455X (gfx1250) — matrix operand order repaired and checked by reading, not compiled, not run on hardware
//
#include <hip/hip_runtime.h>
#include <hip/hip_bf16.h>
#include <stdint.h>

typedef __attribute__((ext_vector_type(16))) __bf16 bf16x16;
typedef __attribute__((ext_vector_type(8)))  float  f32x8;
typedef unsigned int uint32x4 __attribute__((ext_vector_type(4)));
typedef int int32x4 __attribute__((ext_vector_type(4)));
typedef int int32x8 __attribute__((ext_vector_type(8)));

union FA { bf16x16 v; uint32_t u[8]; };

#if __has_builtin(__builtin_amdgcn_s_wait_tensorcnt)
#define TDM_WAIT(n) __builtin_amdgcn_s_wait_tensorcnt(n)
#else
#define TDM_WAIT(n) do { } while (0)
#endif

__device__ __forceinline__ uint16_t f2bf(float f) {
  union { float f; uint32_t u; } x; x.f = f;
  uint32_t u = x.u + 0x7FFFu + ((x.u >> 16) & 1u);
  return (uint16_t)(u >> 16);
}

// raw v_exp_f32: args are <= 0; flush-to-zero below -126 is fine for softmax
__device__ __forceinline__ float fexp2(float x) {
#if __has_builtin(__builtin_amdgcn_exp2f)
  return __builtin_amdgcn_exp2f(x);
#else
  return exp2f(x);
#endif
}
__device__ __forceinline__ float frcp(float x) {
#if __has_builtin(__builtin_amdgcn_rcpf)
  return __builtin_amdgcn_rcpf(x);
#else
  return 1.0f / x;
#endif
}

// ---- TDM: DMA a strided 2D dword tile global->LDS with padded pitch ----
// d0: dwords per row, rows: #rows, stride_dw: global row stride (dwords),
// pic/pac: D# pad_interval / pad_amount codes, pitch_dw: LDS row pitch (fallback).
// Call from ONE wave only; complete with TDM_WAIT + workgroup barrier.
__device__ __forceinline__ void tdm_load_2d(uint32_t* lds, const void* g,
                                            uint32_t d0, uint32_t rows,
                                            uint32_t stride_dw,
                                            uint32_t pic, uint32_t pac,
                                            uint32_t pitch_dw) {
#if __has_builtin(__builtin_amdgcn_tensor_load_to_lds)
  uint64_t ga = (uint64_t)(uintptr_t)g;
  uint32x4 g0;
  g0.x = 1u;                                              // count=1, user D#
  g0.y = (uint32_t)(uintptr_t)lds;                        // LDS byte address
  g0.z = (uint32_t)ga;                                    // global addr [31:0]
  g0.w = ((uint32_t)(ga >> 32) & 0x01FFFFFFu) | 0x80000000u;  // [56:32] | type=2
  int32x8 g1;
  g1[0] = (int)((2u << 16) | (1u << 20) | (pic << 22) | (pac << 25)); // 4B, pad_en
  g1[1] = (int)(d0 << 16);                                // tensor_dim0 lo16
  g1[2] = (int)(((d0 >> 16) & 0xFFFFu) | (rows << 16));   // dim0 hi | dim1 lo
  g1[3] = (int)(((rows >> 16) & 0xFFFFu) | ((d0 & 0xFFFFu) << 16)); // dim1 hi | tile_dim0
  g1[4] = (int)(rows & 0xFFFFu);                          // tile_dim1 (tile_dim2=0)
  g1[5] = (int)stride_dw;                                 // tensor_dim0_stride lo32
  g1[6] = 0;
  g1[7] = 0;
  int32x4 z4 = {0, 0, 0, 0};
#if defined(__clang_major__) && (__clang_major__ >= 23)
  int32x8 z8 = {0, 0, 0, 0, 0, 0, 0, 0};
  __builtin_amdgcn_tensor_load_to_lds(g0, g1, z4, z4, z8, 0);
#else
  __builtin_amdgcn_tensor_load_to_lds(g0, g1, z4, z4, 0);
#endif
#else
  // fallback: calling wave's lanes copy
  const uint32_t* gg = (const uint32_t*)g;
  int lane = (int)(threadIdx.x & 31);
  for (uint32_t r = 0; r < rows; r++)
    for (uint32_t j = lane; j < d0; j += 32)
      lds[r * pitch_dw + j] = gg[(size_t)r * stride_dw + j];
#endif
}

// ---------------- fp32 -> bf16 convert ----------------
__global__ void cvt_bf16_kernel(const float* __restrict__ in,
                                uint16_t* __restrict__ out, int n) {
  int i = blockIdx.x * blockDim.x + threadIdx.x;
  if (i < n) out[i] = f2bf(in[i]);
}

// ---------------- GEMM: C[M,N] = A[M,K] * W[N,K]^T + bias ----------------
// TDM-staged, double-buffered LDS. MODE 0: bf16 head-split out. MODE 1: fp32 +resid.
template<int MODE>
__global__ __launch_bounds__(256)
void gemm_bt_kernel(const uint32_t* __restrict__ A,
                    const uint32_t* __restrict__ W,
                    const float* __restrict__ bias,
                    const float* __restrict__ resid,
                    void* __restrict__ outp,
                    int M, int N, int K) {
  __shared__ uint32_t At[2][128 * 17];   // 128 rows x 32 bf16 (pitch 34 = 17 u32)
  __shared__ uint32_t Wt[2][64 * 17];
  const int K2 = K >> 1;
  const int nk = K2 >> 4;                // K/32 steps
  const int tid = threadIdx.x;
  const int lane = tid & 31, wid = tid >> 5;
  const int wm = wid & 3, wn = wid >> 2; // 4 x 2 wave grid
  const int half = lane >> 4, nl = lane & 15;
  const int m0 = blockIdx.y * 128;
  const int n0 = blockIdx.x * 64;

  const uint32_t* Ab = A + (size_t)m0 * K2;
  const uint32_t* Wb = W + (size_t)n0 * K2;

  f32x8 acc[2][2] = {};

  if (wid == 0) {   // prologue: DMA tile 0
    tdm_load_2d(&At[0][0], Ab, 16, 128, (uint32_t)K2, 3, 0, 17);
    tdm_load_2d(&Wt[0][0], Wb, 16, 64,  (uint32_t)K2, 3, 0, 17);
    TDM_WAIT(0);
  }
  __syncthreads();

  for (int kt = 0; kt < nk; kt++) {
    const int cur = kt & 1;
    if (wid == 0 && kt + 1 < nk) {   // DMA next tile while computing this one
      tdm_load_2d(&At[cur ^ 1][0], Ab + (kt + 1) * 16, 16, 128, (uint32_t)K2, 3, 0, 17);
      tdm_load_2d(&Wt[cur ^ 1][0], Wb + (kt + 1) * 16, 16, 64,  (uint32_t)K2, 3, 0, 17);
    }

    FA a[2], b[2];
#pragma unroll
    for (int mi = 0; mi < 2; mi++) {
      const int mrow = wm * 32 + mi * 16 + nl;
#pragma unroll
      for (int v = 0; v < 8; v++) {
        int kk = (v < 4) ? (8 * half + 2 * v) : (16 + 8 * half + 2 * (v - 4));
        a[mi].u[v] = At[cur][mrow * 17 + (kk >> 1)];
      }
    }
#pragma unroll
    for (int ni = 0; ni < 2; ni++) {
      const int nrow = wn * 32 + ni * 16 + nl;
#pragma unroll
      for (int v = 0; v < 8; v++) {
        int kk = (v < 4) ? (8 * half + 2 * v) : (16 + 8 * half + 2 * (v - 4));
        b[ni].u[v] = Wt[cur][nrow * 17 + (kk >> 1)];
      }
    }
#pragma unroll
    for (int mi = 0; mi < 2; mi++)
#pragma unroll
      for (int ni = 0; ni < 2; ni++)
        acc[mi][ni] = __builtin_amdgcn_wmma_f32_16x16x32_bf16(
            false, a[mi].v, false, b[ni].v, (short)0, acc[mi][ni], false, false);

    if (wid == 0) TDM_WAIT(0);   // next tile landed
    __syncthreads();
  }

#pragma unroll
  for (int mi = 0; mi < 2; mi++)
#pragma unroll
    for (int ni = 0; ni < 2; ni++)
#pragma unroll
      for (int r = 0; r < 8; r++) {
        int m = m0 + wm * 32 + mi * 16 + r + half * 8;
        int n = n0 + wn * 32 + ni * 16 + nl;
        float val = acc[mi][ni][r] + bias[n];
        if (MODE == 0) {   // head-split bf16: [(b*16+h)*2048 + s]*64 + d
          int bb = m >> 11, s = m & 2047, hh = n >> 6, d = n & 63;
          ((uint16_t*)outp)[(((size_t)(bb * 16 + hh) * 2048 + s) << 6) + d] = f2bf(val);
        } else {
          ((float*)outp)[(size_t)m * N + n] = val + resid[(size_t)m * N + n];
        }
      }
}

// ---------------- Attention: scores+softmax(weights out)+ctx ----------------
// grid (Sq/64, B*H), 128 threads (4 waves); wave w owns q-rows [w*16, w*16+16)
// K tiles double-buffered via TDM with pipelined TENSORcnt waits.
__global__ __launch_bounds__(128)
void attn_kernel(const uint32_t* __restrict__ Qb,   // [B*H,2048,32] u32 bf16 pairs
                 const uint32_t* __restrict__ Kb,
                 const uint32_t* __restrict__ Vb,
                 float* __restrict__ attn,          // [B*H,2048,2048] fp32
                 uint16_t* __restrict__ ctxb) {     // [B,2048,1024] bf16
  __shared__ uint32_t Qt[64 * 34];        // 64 x 64 bf16, pitch 68 (34 u32)
  __shared__ uint32_t Kt[2][128 * 34];    // double-buffered 128 keys x 64 bf16
  __shared__ uint32_t Vt[64 * 65];        // transposed: 64 dims x 128 keys, pitch 130
  __shared__ uint32_t Pb[4][16 * 17];     // per-wave P tile 16 x 32 bf16

  const int tid = threadIdx.x;
  const int lane = tid & 31, wid = tid >> 5;
  const int half = lane >> 4, nl = lane & 15;
  const int q0 = blockIdx.x * 64;
  const int bh = blockIdx.y;

  const uint32_t* Qg = Qb + ((size_t)bh * 2048 + q0) * 32;
  const uint32_t* Kg = Kb + (size_t)bh * 2048 * 32;
  const uint32_t* Vg = Vb + (size_t)bh * 2048 * 32;

  if (wid == 0) {   // Q tile + first K tile DMA; wait only for Q (in-order TDM)
    tdm_load_2d(Qt, Qg, 32, 64, 32, 4, 1, 34);
    tdm_load_2d(Kt[0], Kg, 32, 128, 32, 4, 1, 34);
    TDM_WAIT(1);
  }
  __syncthreads();

  FA aq[2];   // persistent Q fragments (k = 0..31, 32..63)
  {
    int m = wid * 16 + nl;
#pragma unroll
    for (int s = 0; s < 2; s++)
#pragma unroll
      for (int v = 0; v < 8; v++) {
        int kk = 32 * s + ((v < 4) ? (8 * half + 2 * v) : (16 + 8 * half + 2 * (v - 4)));
        aq[s].u[v] = Qt[m * 34 + (kk >> 1)];
      }
  }

  // 1/sqrt(1024) folded with log2(e): softmax in base-2 domain
  const float scl2 = 0.03125f * 1.44269504088896340736f;
  float mrow[8], lrow[8];
#pragma unroll
  for (int r = 0; r < 8; r++) { mrow[r] = -3.0e38f; lrow[r] = 0.f; }

  // ---- pass 1: per-lane online softmax partials ----
  for (int kt = 0; kt < 16; kt++) {
    const int cur = kt & 1;
    if (wid == 0) {
      if (kt + 1 < 16) {   // prefetch next K tile, wait only for current
        tdm_load_2d(Kt[cur ^ 1], Kg + (size_t)(kt + 1) * 4096, 32, 128, 32, 4, 1, 34);
        TDM_WAIT(1);
      } else {
        TDM_WAIT(0);
      }
    }
    __syncthreads();
    for (int nb = 0; nb < 8; nb++) {
      FA bk[2];
#pragma unroll
      for (int s = 0; s < 2; s++)
#pragma unroll
        for (int v = 0; v < 8; v++) {
          int kk = 32 * s + ((v < 4) ? (8 * half + 2 * v) : (16 + 8 * half + 2 * (v - 4)));
          bk[s].u[v] = Kt[cur][(nb * 16 + nl) * 34 + (kk >> 1)];
        }
      f32x8 sc = {};
      sc = __builtin_amdgcn_wmma_f32_16x16x32_bf16(false, aq[0].v, false, bk[0].v, (short)0, sc, false, false);
      sc = __builtin_amdgcn_wmma_f32_16x16x32_bf16(false, aq[1].v, false, bk[1].v, (short)0, sc, false, false);
#pragma unroll
      for (int r = 0; r < 8; r++) {
        float s2 = sc[r] * scl2;
        float mnew = fmaxf(mrow[r], s2);
        lrow[r] = lrow[r] * fexp2(mrow[r] - mnew) + fexp2(s2 - mnew);
        mrow[r] = mnew;
      }
    }
    __syncthreads();
  }
  // ---- single cross-lane merge of (m,l) partials ----
#pragma unroll
  for (int r = 0; r < 8; r++) {
    float m = mrow[r], l = lrow[r];
#pragma unroll
    for (int mk = 1; mk < 16; mk <<= 1) {
      float mo = __shfl_xor(m, mk, 32);
      float lo = __shfl_xor(l, mk, 32);
      float mn = fmaxf(m, mo);
      l = l * fexp2(m - mn) + lo * fexp2(mo - mn);
      m = mn;
    }
    mrow[r] = m;
    lrow[r] = frcp(l);
  }

  // ---- pass 2: write weights, accumulate ctx ----
  f32x8 cacc[4] = {};
  float* attn_base = attn + ((size_t)bh * 2048 + q0) * 2048;
  uint16_t* pb = (uint16_t*)&Pb[wid][0];
  if (wid == 0) tdm_load_2d(Kt[0], Kg, 32, 128, 32, 4, 1, 34);  // re-prime K0
  for (int kt = 0; kt < 16; kt++) {
    const int cur = kt & 1;
    {   // stage V transposed (overlaps with in-flight K DMA)
      uint16_t* vt = (uint16_t*)Vt;
#pragma unroll
      for (int j = 0; j < 32; j++) {
        uint32_t v = Vg[(size_t)(kt * 128 + tid) * 32 + j];
        vt[(2 * j) * 130 + tid]     = (uint16_t)v;
        vt[(2 * j + 1) * 130 + tid] = (uint16_t)(v >> 16);
      }
    }
    if (wid == 0) {
      if (kt + 1 < 16) {
        tdm_load_2d(Kt[cur ^ 1], Kg + (size_t)(kt + 1) * 4096, 32, 128, 32, 4, 1, 34);
        TDM_WAIT(1);
      } else {
        TDM_WAIT(0);
      }
    }
    __syncthreads();
    for (int ch = 0; ch < 4; ch++) {        // 32-key chunks
      int kc = ch * 32;
#pragma unroll
      for (int sub = 0; sub < 2; sub++) {
        int key0 = kc + sub * 16;
        FA bk[2];
#pragma unroll
        for (int s = 0; s < 2; s++)
#pragma unroll
          for (int v = 0; v < 8; v++) {
            int kk = 32 * s + ((v < 4) ? (8 * half + 2 * v) : (16 + 8 * half + 2 * (v - 4)));
            bk[s].u[v] = Kt[cur][(key0 + nl) * 34 + (kk >> 1)];
          }
        f32x8 sc = {};
        sc = __builtin_amdgcn_wmma_f32_16x16x32_bf16(false, aq[0].v, false, bk[0].v, (short)0, sc, false, false);
        sc = __builtin_amdgcn_wmma_f32_16x16x32_bf16(false, aq[1].v, false, bk[1].v, (short)0, sc, false, false);
#pragma unroll
        for (int r = 0; r < 8; r++) {
          float p = fexp2(sc[r] * scl2 - mrow[r]) * lrow[r];
          int mloc = r + 8 * half;
          attn_base[(size_t)(wid * 16 + mloc) * 2048 + kt * 128 + key0 + nl] = p;
          pb[mloc * 34 + sub * 16 + nl] = f2bf(p);
        }
      }
      asm volatile("s_wait_dscnt 0x0" ::: "memory");   // P writes -> P reads (same wave)
      FA pf;
#pragma unroll
      for (int v = 0; v < 8; v++) {
        int kk = (v < 4) ? (8 * half + 2 * v) : (16 + 8 * half + 2 * (v - 4));
        pf.u[v] = Pb[wid][nl * 17 + (kk >> 1)];
      }
#pragma unroll
      for (int db = 0; db < 4; db++) {
        FA vf;
        int d = db * 16 + nl;
#pragma unroll
        for (int v = 0; v < 8; v++) {
          int key = kc + ((v < 4) ? (8 * half + 2 * v) : (16 + 8 * half + 2 * (v - 4)));
          vf.u[v] = Vt[d * 65 + (key >> 1)];
        }
        cacc[db] = __builtin_amdgcn_wmma_f32_16x16x32_bf16(
            false, pf.v, false, vf.v, (short)0, cacc[db], false, false);
      }
    }
    __syncthreads();
  }

  int b = bh >> 4, h = bh & 15;
#pragma unroll
  for (int db = 0; db < 4; db++)
#pragma unroll
    for (int r = 0; r < 8; r++) {
      int qrow = q0 + wid * 16 + r + half * 8;
      int d = h * 64 + db * 16 + nl;
      ctxb[((size_t)(b * 2048 + qrow) << 10) + d] = f2bf(cacc[db][r]);
    }
}

// ---------------- LayerNorm ----------------
__global__ __launch_bounds__(256)
void ln_kernel(const float* __restrict__ y, const float* __restrict__ gamma,
               const float* __restrict__ beta, float* __restrict__ out) {
  __shared__ float rs[8], rss[8];
  int row = blockIdx.x;
  const float* yr = y + (size_t)row * 1024;
  float v[4], s = 0.f, ss = 0.f;
#pragma unroll
  for (int i = 0; i < 4; i++) {
    v[i] = yr[threadIdx.x * 4 + i];
    s += v[i]; ss += v[i] * v[i];
  }
#pragma unroll
  for (int mk = 1; mk < 32; mk <<= 1) {
    s += __shfl_xor(s, mk, 32); ss += __shfl_xor(ss, mk, 32);
  }
  if ((threadIdx.x & 31) == 0) { rs[threadIdx.x >> 5] = s; rss[threadIdx.x >> 5] = ss; }
  __syncthreads();
  s = 0.f; ss = 0.f;
#pragma unroll
  for (int w = 0; w < 8; w++) { s += rs[w]; ss += rss[w]; }
  float mu = s * (1.0f / 1024.0f);
  float var = ss * (1.0f / 1024.0f) - mu * mu;
  float rstd = rsqrtf(var + 1e-5f);
  float* orow = out + (size_t)row * 1024;
#pragma unroll
  for (int i = 0; i < 4; i++) {
    int c = threadIdx.x * 4 + i;
    orow[c] = (v[i] - mu) * rstd * gamma[c] + beta[c];
  }
}

extern "C" void kernel_launch(void* const* d_in, const int* in_sizes, int n_in,
                              void* d_out, int out_size, void* d_ws, size_t ws_size,
                              hipStream_t stream) {
  (void)in_sizes; (void)n_in; (void)out_size; (void)ws_size;
  const float* x   = (const float*)d_in[0];
  const float* enc = (const float*)d_in[1];
  const float* Wq  = (const float*)d_in[2];
  const float* bq  = (const float*)d_in[3];
  const float* Wk  = (const float*)d_in[4];
  const float* bk  = (const float*)d_in[5];
  const float* Wv  = (const float*)d_in[6];
  const float* bv  = (const float*)d_in[7];
  const float* Wo  = (const float*)d_in[8];
  const float* bo  = (const float*)d_in[9];
  const float* gamma = (const float*)d_in[10];
  const float* beta  = (const float*)d_in[11];

  const int M = 4096, N = 1024, K = 1024;          // (B*Sq), D, D
  const size_t MB = 1u << 20;
  char* ws = (char*)d_ws;
  uint16_t* xb   = (uint16_t*)(ws + 0 * MB);
  uint16_t* eb   = (uint16_t*)(ws + 8 * MB);
  uint16_t* Wqb  = (uint16_t*)(ws + 16 * MB);
  uint16_t* Wkb  = (uint16_t*)(ws + 18 * MB);
  uint16_t* Wvb  = (uint16_t*)(ws + 20 * MB);
  uint16_t* Wob  = (uint16_t*)(ws + 22 * MB);
  uint16_t* Qbf  = (uint16_t*)(ws + 24 * MB);
  uint16_t* Kbf  = (uint16_t*)(ws + 32 * MB);
  uint16_t* Vbf  = (uint16_t*)(ws + 40 * MB);
  uint16_t* ctxb = (uint16_t*)(ws + 48 * MB);
  float*    yb   = (float*)   (ws + 56 * MB);

  float* out_ln   = (float*)d_out;                  // [2,2048,1024]
  float* attn_out = (float*)d_out + 4194304;        // [2,16,2048,2048]

  // 1. converts
  cvt_bf16_kernel<<<16384, 256, 0, stream>>>(x,   xb,  4194304);
  cvt_bf16_kernel<<<16384, 256, 0, stream>>>(enc, eb,  4194304);
  cvt_bf16_kernel<<<4096,  256, 0, stream>>>(Wq,  Wqb, 1048576);
  cvt_bf16_kernel<<<4096,  256, 0, stream>>>(Wk,  Wkb, 1048576);
  cvt_bf16_kernel<<<4096,  256, 0, stream>>>(Wv,  Wvb, 1048576);
  cvt_bf16_kernel<<<4096,  256, 0, stream>>>(Wo,  Wob, 1048576);

  // 2. Q/K/V projections (head-split bf16 output)
  dim3 ggrid(N / 64, M / 128);
  gemm_bt_kernel<0><<<ggrid, 256, 0, stream>>>((const uint32_t*)xb, (const uint32_t*)Wqb, bq, nullptr, Qbf, M, N, K);
  gemm_bt_kernel<0><<<ggrid, 256, 0, stream>>>((const uint32_t*)eb, (const uint32_t*)Wkb, bk, nullptr, Kbf, M, N, K);
  gemm_bt_kernel<0><<<ggrid, 256, 0, stream>>>((const uint32_t*)eb, (const uint32_t*)Wvb, bv, nullptr, Vbf, M, N, K);

  // 3. attention (writes softmax weights + bf16 ctx)
  attn_kernel<<<dim3(32, 32), 128, 0, stream>>>((const uint32_t*)Qbf, (const uint32_t*)Kbf,
                                                (const uint32_t*)Vbf, attn_out, ctxb);

  // 4. output projection + residual
  gemm_bt_kernel<1><<<ggrid, 256, 0, stream>>>((const uint32_t*)ctxb, (const uint32_t*)Wob, bo, x, yb, M, N, K);

  // 5. layernorm
  ln_kernel<<<4096, 256, 0, stream>>>(yb, gamma, beta, out_ln);
}
